// SeqDecoder_42339787604238
// MI455X (gfx1250) — compile-verified, matrix operand order checked by reading, not run on hardware
//
#include <hip/hip_runtime.h>

#define B_    4096
#define T_    256
#define COND_ 128
#define INTV_ 64
#define H_    256
#define MTILE 16      // batch rows per workgroup
#define NTHREADS 256  // 8 waves: 2 waves/SIMD -> 512-VGPR budget, weights stay register-resident

typedef __attribute__((ext_vector_type(16))) __bf16 bf16x16;
typedef __attribute__((ext_vector_type(8)))  float  f32x8;

union Frag { bf16x16 v; unsigned short u[16]; uint4 q[2]; };

__device__ __forceinline__ unsigned short f2bf(float f) {
  union { float f; unsigned u; } c; c.f = f;
  unsigned r = c.u + 0x7FFFu + ((c.u >> 16) & 1u);   // round-to-nearest-even
  return (unsigned short)(r >> 16);
}

__global__ __launch_bounds__(NTHREADS, 1) __attribute__((amdgpu_waves_per_eu(2)))
void seq_decoder_gru(
    const float* __restrict__ cond,  const float* __restrict__ intv,
    const float* __restrict__ times,
    const float* __restrict__ w1,    const float* __restrict__ b1,
    const float* __restrict__ w2,    const float* __restrict__ b2,
    const float* __restrict__ w3,    const float* __restrict__ b3,
    const float* __restrict__ w_h0,  const float* __restrict__ b_h0,
    const float* __restrict__ w_ih,  const float* __restrict__ b_ih,
    const float* __restrict__ w_hh,  const float* __restrict__ b_hh,
    const float* __restrict__ w_head,const float* __restrict__ b_head,
    float* __restrict__ out)
{
  __shared__ float sA[MTILE * 128];                        // s1, then cond tile
  __shared__ float sB[MTILE * 128];                        // s2
  __shared__ float sC[MTILE * 64];                         // interval tile, then seq_emb
  __shared__ __align__(16) unsigned short hbf[MTILE * H_]; // h_t as bf16 (A-matrix source)
  __shared__ __align__(16) float xbuf[MTILE];              // per-row scalar GRU input
  __shared__ float head_buf[2][8][MTILE][2];               // double-buffered head partials

  const int tid  = threadIdx.x;
  const int wv   = tid >> 5;
  const int lane = tid & 31;
  const int half = lane >> 4;
  const int ln   = lane & 15;
  const int n0   = wv * 32 + ln;                 // first hidden column owned by this lane
  const int n1   = n0 + 16;                      // second hidden column
  const int m0   = blockIdx.x * MTILE;

  // ---------------- Phase 0: fused seq-feature MLP (tiny VALU prologue) -------------------
  for (int idx = tid; idx < MTILE * INTV_; idx += NTHREADS) {
    int r = idx >> 6, c = idx & 63;
    sC[idx] = intv[(m0 + r) * INTV_ + c];
  }
  __syncthreads();
  for (int idx = tid; idx < MTILE * 128; idx += NTHREADS) {   // s1 = relu(x@w1+b1)
    int r = idx >> 7, j = idx & 127;
    float acc = b1[j];
    for (int k = 0; k < INTV_; ++k) acc += sC[r * 64 + k] * w1[k * 128 + j];
    sA[idx] = fmaxf(acc, 0.f);
  }
  __syncthreads();
  for (int idx = tid; idx < MTILE * 128; idx += NTHREADS) {   // s2 = relu(s1@w2+b2)
    int r = idx >> 7, j = idx & 127;
    float acc = b2[j];
    for (int k = 0; k < 128; ++k) acc += sA[r * 128 + k] * w2[k * 128 + j];
    sB[idx] = fmaxf(acc, 0.f);
  }
  __syncthreads();
  for (int idx = tid; idx < MTILE * INTV_; idx += NTHREADS) { // seq_emb = s2@w3+b3
    int r = idx >> 6, j = idx & 63;
    float acc = b3[j];
    for (int k = 0; k < 128; ++k) acc += sB[r * 128 + k] * w3[k * 64 + j];
    sC[idx] = acc;
  }
  for (int idx = tid; idx < MTILE * COND_; idx += NTHREADS) { // cond tile (sA free now)
    int r = idx >> 7, c = idx & 127;
    sA[idx] = cond[(m0 + r) * COND_ + c];
  }
  __syncthreads();

  // h0 directly in WMMA C-fragment layout: lane holds (rows i+8*half, cols n0 and n1)
  float h0v[8], h1v[8];
  #pragma unroll
  for (int i = 0; i < 8; ++i) {
    int row = i + half * 8;
    float a0 = b_h0[n0], a1 = b_h0[n1];
    for (int k = 0; k < COND_; ++k) {
      float v = sA[row * 128 + k];
      a0 += v * w_h0[k * H_ + n0];
      a1 += v * w_h0[k * H_ + n1];
    }
    for (int k = 0; k < INTV_; ++k) {
      float v = sC[row * 64 + k];
      a0 += v * w_h0[(COND_ + k) * H_ + n0];
      a1 += v * w_h0[(COND_ + k) * H_ + n1];
    }
    h0v[i] = tanhf(a0);
    h1v[i] = tanhf(a1);
  }

  // --------- Weight-stationary B-fragments: 6 tiles (r,z,n for 2 column groups) -----------
  // B layout (mirrors ISA 7.12.2 A): lanes 0-15 K runs {0..7,16..23}; lanes 16-31 {8..15,24..31}
  bf16x16 Bw[6][8];
  #pragma unroll
  for (int kc = 0; kc < 8; ++kc) {
    Frag f0, f1, f2, f3, f4, f5;
    #pragma unroll
    for (int e = 0; e < 16; ++e) {
      int koff = (e < 8) ? (half * 8 + e) : (8 + half * 8 + e);
      int K = kc * 32 + koff;
      f0.u[e] = f2bf(w_hh[K * 768 + n0]);
      f1.u[e] = f2bf(w_hh[K * 768 + H_ + n0]);
      f2.u[e] = f2bf(w_hh[K * 768 + 2 * H_ + n0]);
      f3.u[e] = f2bf(w_hh[K * 768 + n1]);
      f4.u[e] = f2bf(w_hh[K * 768 + H_ + n1]);
      f5.u[e] = f2bf(w_hh[K * 768 + 2 * H_ + n1]);
    }
    Bw[0][kc] = f0.v; Bw[1][kc] = f1.v; Bw[2][kc] = f2.v;
    Bw[3][kc] = f3.v; Bw[4][kc] = f4.v; Bw[5][kc] = f5.v;
  }

  // Per-lane gate constants (input bias folded into hidden bias for r,z)
  const float wir0 = w_ih[n0], wiz0 = w_ih[H_ + n0], win0 = w_ih[2 * H_ + n0];
  const float brc0 = b_ih[n0] + b_hh[n0], bzc0 = b_ih[H_ + n0] + b_hh[H_ + n0];
  const float bin0 = b_ih[2 * H_ + n0], bhn0 = b_hh[2 * H_ + n0];
  const float wir1 = w_ih[n1], wiz1 = w_ih[H_ + n1], win1 = w_ih[2 * H_ + n1];
  const float brc1 = b_ih[n1] + b_hh[n1], bzc1 = b_ih[H_ + n1] + b_hh[H_ + n1];
  const float bin1 = b_ih[2 * H_ + n1], bhn1 = b_hh[2 * H_ + n1];
  const float wh00 = w_head[n0 * 2], wh01 = w_head[n0 * 2 + 1];
  const float wh10 = w_head[n1 * 2], wh11 = w_head[n1 * 2 + 1];
  const float bhd  = b_head[lane & 1];
  const long long OB = (long long)B_ * (T_ + 1);

  // ---------------- Recurrent loop: 257 steps ---------------------------------------------
  float prev = 0.f;
  #pragma unroll 1
  for (int t = 0; t <= T_; ++t) {
    #pragma unroll
    for (int i = 0; i < 8; ++i) {                        // publish h_t as bf16 A-matrix
      hbf[(i + half * 8) * H_ + n0] = f2bf(h0v[i]);
      hbf[(i + half * 8) * H_ + n1] = f2bf(h1v[i]);
    }
    if (wv == 1 && lane < 16) {                          // scalar GRU input x_t per row
      float x = 0.f;
      if (t > 0) { float cur = times[(m0 + ln) * T_ + (t - 1)]; x = cur - prev; prev = cur; }
      xbuf[ln] = x;
    }
    __syncthreads();                                     // B1: h_t + x_t published

    if (wv == 0 && t > 0) {                              // finalize out_{t-1} from partials
      int row = lane >> 1, o = lane & 1;
      float s = bhd;
      #pragma unroll
      for (int w2 = 0; w2 < 8; ++w2) s += head_buf[(t - 1) & 1][w2][row][o];
      out[(long long)o * OB + (long long)(m0 + row) * (T_ + 1) + (t - 1)] = s;
    }

    f32x8 aR0 = {}, aZ0 = {}, aN0 = {}, aR1 = {}, aZ1 = {}, aN1 = {};
    #pragma unroll
    for (int kc = 0; kc < 8; ++kc) {
      Frag a;                                            // A-frag: lane=row, 2x16B K runs
      const uint4* pa = (const uint4*)((const char*)hbf + ln * 512 + kc * 64 + half * 16);
      a.q[0] = pa[0];
      a.q[1] = pa[2];
      aR0 = __builtin_amdgcn_wmma_f32_16x16x32_bf16(false, a.v, false, Bw[0][kc], (short)0, aR0, false, false);
      aZ0 = __builtin_amdgcn_wmma_f32_16x16x32_bf16(false, a.v, false, Bw[1][kc], (short)0, aZ0, false, false);
      aN0 = __builtin_amdgcn_wmma_f32_16x16x32_bf16(false, a.v, false, Bw[2][kc], (short)0, aN0, false, false);
      aR1 = __builtin_amdgcn_wmma_f32_16x16x32_bf16(false, a.v, false, Bw[3][kc], (short)0, aR1, false, false);
      aZ1 = __builtin_amdgcn_wmma_f32_16x16x32_bf16(false, a.v, false, Bw[4][kc], (short)0, aZ1, false, false);
      aN1 = __builtin_amdgcn_wmma_f32_16x16x32_bf16(false, a.v, false, Bw[5][kc], (short)0, aN1, false, false);
    }

    const float4* xp = (const float4*)&xbuf[half * 8];
    float4 x0 = xp[0], x1 = xp[1];
    float xv[8] = {x0.x, x0.y, x0.z, x0.w, x1.x, x1.y, x1.z, x1.w};

    #pragma unroll
    for (int i = 0; i < 8; ++i) {                        // lane-local gate math (f32)
      float r0 = 1.f / (1.f + __expf(-(aR0[i] + xv[i] * wir0 + brc0)));
      float z0 = 1.f / (1.f + __expf(-(aZ0[i] + xv[i] * wiz0 + bzc0)));
      float nn0 = tanhf(xv[i] * win0 + bin0 + r0 * (aN0[i] + bhn0));
      h0v[i] = (1.f - z0) * nn0 + z0 * h0v[i];
      float r1 = 1.f / (1.f + __expf(-(aR1[i] + xv[i] * wir1 + brc1)));
      float z1 = 1.f / (1.f + __expf(-(aZ1[i] + xv[i] * wiz1 + bzc1)));
      float nn1 = tanhf(xv[i] * win1 + bin1 + r1 * (aN1[i] + bhn1));
      h1v[i] = (1.f - z1) * nn1 + z1 * h1v[i];
    }

    {                                                    // head partials for out_t (h_{t+1})
      float p[8];
      #pragma unroll
      for (int i = 0; i < 8; ++i) p[i] = h0v[i] * wh00 + h1v[i] * wh10;
      #pragma unroll
      for (int m = 1; m < 16; m <<= 1) {
        #pragma unroll
        for (int i = 0; i < 8; ++i) p[i] += __shfl_xor(p[i], m, 16);
      }
      if (ln == 0) {
        #pragma unroll
        for (int i = 0; i < 8; ++i) head_buf[t & 1][wv][half * 8 + i][0] = p[i];
      }
      #pragma unroll
      for (int i = 0; i < 8; ++i) p[i] = h0v[i] * wh01 + h1v[i] * wh11;
      #pragma unroll
      for (int m = 1; m < 16; m <<= 1) {
        #pragma unroll
        for (int i = 0; i < 8; ++i) p[i] += __shfl_xor(p[i], m, 16);
      }
      if (ln == 0) {
        #pragma unroll
        for (int i = 0; i < 8; ++i) head_buf[t & 1][wv][half * 8 + i][1] = p[i];
      }
    }
    __syncthreads();                                     // B2: reads done, partials published
  }

  // epilogue: finalize out_{T} (partials written at t=T with parity 0)
  if (wv == 0) {
    int row = lane >> 1, o = lane & 1;
    float s = bhd;
    #pragma unroll
    for (int w2 = 0; w2 < 8; ++w2) s += head_buf[T_ & 1][w2][row][o];
    out[(long long)o * OB + (long long)(m0 + row) * (T_ + 1) + T_] = s;
  }
}

extern "C" void kernel_launch(void* const* d_in, const int* in_sizes, int n_in,
                              void* d_out, int out_size, void* d_ws, size_t ws_size,
                              hipStream_t stream) {
  (void)in_sizes; (void)n_in; (void)d_ws; (void)ws_size; (void)out_size;
  const float* cond   = (const float*)d_in[0];
  const float* intv   = (const float*)d_in[1];
  const float* times  = (const float*)d_in[2];
  /* d_in[3] = seq_lens (unused by reference outputs) */
  const float* w1     = (const float*)d_in[4];
  const float* b1     = (const float*)d_in[5];
  const float* w2     = (const float*)d_in[6];
  const float* b2     = (const float*)d_in[7];
  const float* w3     = (const float*)d_in[8];
  const float* b3     = (const float*)d_in[9];
  const float* w_h0   = (const float*)d_in[10];
  const float* b_h0   = (const float*)d_in[11];
  const float* w_ih   = (const float*)d_in[12];
  const float* b_ih   = (const float*)d_in[13];
  const float* w_hh   = (const float*)d_in[14];
  const float* b_hh   = (const float*)d_in[15];
  const float* w_head = (const float*)d_in[16];
  const float* b_head = (const float*)d_in[17];
  float* out = (float*)d_out;

  seq_decoder_gru<<<B_ / MTILE, NTHREADS, 0, stream>>>(
      cond, intv, times, w1, b1, w2, b2, w3, b3, w_h0, b_h0,
      w_ih, b_ih, w_hh, b_hh, w_head, b_head, out);
}
